// GATNet_simple_withEdges_44281112822534
// MI455X (gfx1250) — matrix operand order repaired and checked by reading, not compiled, not run on hardware
//
#include <hip/hip_runtime.h>
#include <hip/hip_bf16.h>
#include <cmath>

// ---------------------------------------------------------------------------
// Problem constants (from reference setup_inputs)
// ---------------------------------------------------------------------------
#define NN    20000
#define EE    320000
#define EP    (EE + NN)      // edges + self loops = 340000
#define EDIM  22
#define NEG_SLOPE 0.2f

typedef __attribute__((ext_vector_type(16))) _Float16 v16h;
typedef __attribute__((ext_vector_type(8)))  _Float16 v8h;
typedef __attribute__((ext_vector_type(8)))  float    v8f;

// ---------------------------------------------------------------------------
// Small helpers
// ---------------------------------------------------------------------------
__device__ __forceinline__ float atomicMaxF(float* addr, float val) {
    // IEEE-754 order-preserving trick; addr must be initialized to -inf.
    if (val >= 0.0f)
        return __int_as_float(atomicMax((int*)addr, __float_as_int(val)));
    return __uint_as_float(atomicMin((unsigned int*)addr, __float_as_uint(val)));
}

__global__ void fill_f32(float* __restrict__ p, float v, long long n) {
    long long i = (long long)blockIdx.x * blockDim.x + threadIdx.x;
    long long stride = (long long)gridDim.x * blockDim.x;
    for (; i < n; i += stride) p[i] = v;
}

// ---------------------------------------------------------------------------
// Self-loop 'mean' edge attributes
// ---------------------------------------------------------------------------
__global__ void loop_cnt_kernel(const int* __restrict__ dst, float* __restrict__ cnt) {
    int e = blockIdx.x * blockDim.x + threadIdx.x;
    if (e < EE) atomicAdd(&cnt[dst[e]], 1.0f);
}

__global__ void loop_easum_kernel(const int* __restrict__ dst, const float* __restrict__ ea,
                                  float* __restrict__ mea) {
    long long idx = (long long)blockIdx.x * blockDim.x + threadIdx.x;
    if (idx >= (long long)EE * EDIM) return;
    int e = (int)(idx / EDIM);
    int k = (int)(idx % EDIM);
    atomicAdd(&mea[(size_t)dst[e] * EDIM + k], ea[idx]);
}

__global__ void loop_mean_kernel(const float* __restrict__ cnt, float* __restrict__ mea) {
    long long idx = (long long)blockIdx.x * blockDim.x + threadIdx.x;
    if (idx >= (long long)NN * EDIM) return;
    int n = (int)(idx / EDIM);
    mea[idx] = mea[idx] / fmaxf(cnt[n], 1.0f);
}

// ---------------------------------------------------------------------------
// Convert x [N,16] f32 -> [N,32] f16 zero-padded (so all layers use K%32==0)
// ---------------------------------------------------------------------------
__global__ void convert_pad_x_kernel(const float* __restrict__ x, _Float16* __restrict__ xh) {
    long long idx = (long long)blockIdx.x * blockDim.x + threadIdx.x;
    if (idx >= (long long)NN * 32) return;
    int n = (int)(idx >> 5);
    int k = (int)(idx & 31);
    xh[idx] = (k < 16) ? (_Float16)x[n * 16 + k] : (_Float16)0.0f;
}

// ---------------------------------------------------------------------------
// Pack W [Kin,Nc] f32 into the exact B-operand (32x16 K-major) register
// layout consumed by v_wmma_f32_16x16x32_f16, zero-padding K up to Kpad.
// Layout per ISA 7.12.2: lanes 0-15 hold K=kt*32+0..7,16..23 for N=n0+lane,
// lanes 16-31 hold K=kt*32+8..15,24..31 for N=n0+(lane-16).
// Packed index: (((kt*ntiles)+nt)*32 + lane)*16 + i
// ---------------------------------------------------------------------------
__global__ void pack_w_kernel(const float* __restrict__ W, _Float16* __restrict__ Bp,
                              int Kin, int Kpad, int Nc) {
    int ntiles = Nc >> 4;
    long long total = (long long)(Kpad >> 5) * ntiles * 512;
    long long idx = (long long)blockIdx.x * blockDim.x + threadIdx.x;
    if (idx >= total) return;
    int i    = (int)(idx & 15);
    int lane = (int)((idx >> 4) & 31);
    long long t = idx >> 9;
    int nt = (int)(t % ntiles);
    int kt = (int)(t / ntiles);
    int n = nt * 16 + (lane & 15);
    int k = kt * 32 + ((lane >> 4) ? 8 : 0) + i + ((i >= 8) ? 8 : 0);
    Bp[idx] = (k < Kin) ? (_Float16)W[(size_t)k * Nc + n] : (_Float16)0.0f;
}

// ---------------------------------------------------------------------------
// WMMA GEMM: C[M,Nc](f16) = A[M,K](f16 row-major) x Bpacked, f32 accumulate.
// One wave per 16x16 tile, K stepped by 32 via v_wmma_f32_16x16x32_f16.
// A-operand layout (16-bit A 16x32): lane<16 -> row M=lane, halves K=0..7
// and K=16..23; lane>=16 -> row M=lane-16, halves K=8..15 and K=24..31.
// ---------------------------------------------------------------------------
__global__ __launch_bounds__(32)
void gemm_wmma_f16(const _Float16* __restrict__ A, const _Float16* __restrict__ Bp,
                   _Float16* __restrict__ C, int K, int Nc) {
    const int lane = threadIdx.x & 31;
    const int r    = lane & 15;
    const int hi   = lane >> 4;
    const int m0   = blockIdx.x << 4;
    const int nt   = blockIdx.y;
    const int ntiles = Nc >> 4;
    const int ktiles = K >> 5;

    const _Float16* arow = A + (size_t)(m0 + r) * K + hi * 8;
    const _Float16* bp   = Bp + ((size_t)nt * 32 + lane) * 16;

    v8f acc = {};
    for (int kt = 0; kt < ktiles; ++kt) {
        v8h alo = *(const v8h*)(arow + kt * 32);
        v8h ahi = *(const v8h*)(arow + kt * 32 + 16);
        v16h a, b;
        b = *(const v16h*)(bp + (size_t)kt * ntiles * 512);
#pragma unroll
        for (int i = 0; i < 8; ++i) { a[i] = alo[i]; a[8 + i] = ahi[i]; }
        acc = __builtin_amdgcn_wmma_f32_16x16x32_f16(
            /*neg_a=*/false, a, /*neg_b=*/false, b,
            /*c_mod=*/(short)0, acc, /*reuse_a=*/false, /*reuse_b=*/false);
    }
#pragma unroll
    for (int i = 0; i < 8; ++i) {
        // D layout: VGPR i -> M = i (+8 for upper half-wave), N = lane&15
        C[(size_t)(m0 + i + hi * 8) * Nc + (nt * 16 + r)] = (_Float16)acc[i];
    }
}

// ---------------------------------------------------------------------------
// a_s[n,h] = sum_c h[n,h,c]*att_s[h,c] ; same for a_d
// ---------------------------------------------------------------------------
__global__ void attn_node_kernel(const _Float16* __restrict__ h,
                                 const float* __restrict__ atts, const float* __restrict__ attd,
                                 float* __restrict__ as_, float* __restrict__ ad_,
                                 int H, int C) {
    int idx = blockIdx.x * blockDim.x + threadIdx.x;
    if (idx >= NN * H) return;
    int n = idx / H, hh = idx % H;
    const _Float16* hr = h + (size_t)n * H * C + hh * C;
    float s = 0.0f, d = 0.0f;
    for (int c = 0; c < C; ++c) {
        float v = (float)hr[c];
        s += v * atts[hh * C + c];
        d += v * attd[hh * C + c];
    }
    as_[idx] = s;
    ad_[idx] = d;
}

// ---------------------------------------------------------------------------
// Fold We (ED x H*C) with att_e (H x C): wea[k,h] = sum_c We[k,h*C+c]*att_e[h,c]
// ---------------------------------------------------------------------------
__global__ void fold_we_kernel(const float* __restrict__ We, const float* __restrict__ atte,
                               float* __restrict__ wea, int H, int C) {
    int idx = blockIdx.x * blockDim.x + threadIdx.x;
    if (idx >= EDIM * H) return;
    int k = idx / H, hh = idx % H;
    float s = 0.0f;
    for (int c = 0; c < C; ++c) s += We[(size_t)k * H * C + hh * C + c] * atte[hh * C + c];
    wea[idx] = s;
}

// ---------------------------------------------------------------------------
// alpha = leakyrelu(a_s[src] + a_d[dst] + ea . wea) ; amax = segment max
// ---------------------------------------------------------------------------
__global__ void edge_alpha_kernel(const int* __restrict__ src, const int* __restrict__ dst,
                                  const float* __restrict__ ea, const float* __restrict__ mea,
                                  const float* __restrict__ wea,
                                  const float* __restrict__ as_, const float* __restrict__ ad_,
                                  float* __restrict__ alpha, float* __restrict__ amax, int H) {
    long long idx = (long long)blockIdx.x * blockDim.x + threadIdx.x;
    if (idx >= (long long)EP * H) return;
    int e = (int)(idx / H), hh = (int)(idx % H);
    int s, d;
    const float* er;
    if (e < EE) { s = src[e]; d = dst[e]; er = ea + (size_t)e * EDIM; }
    else        { s = d = e - EE;         er = mea + (size_t)(e - EE) * EDIM; }
    float ae = 0.0f;
#pragma unroll
    for (int k = 0; k < EDIM; ++k) ae += er[k] * wea[k * H + hh];
    float v = as_[s * H + hh] + ad_[d * H + hh] + ae;
    v = (v > 0.0f) ? v : NEG_SLOPE * v;
    alpha[idx] = v;
    atomicMaxF(&amax[d * H + hh], v);
}

__global__ void edge_exp_kernel(const int* __restrict__ dst, const float* __restrict__ amax,
                                float* __restrict__ alpha, float* __restrict__ denom, int H) {
    long long idx = (long long)blockIdx.x * blockDim.x + threadIdx.x;
    if (idx >= (long long)EP * H) return;
    int e = (int)(idx / H), hh = (int)(idx % H);
    int d = (e < EE) ? dst[e] : (e - EE);
    float ex = expf(alpha[idx] - amax[d * H + hh]);
    alpha[idx] = ex;
    atomicAdd(&denom[d * H + hh], ex);
}

__global__ void edge_norm_kernel(const int* __restrict__ dst, const float* __restrict__ denom,
                                 float* __restrict__ alpha, int H) {
    long long idx = (long long)blockIdx.x * blockDim.x + threadIdx.x;
    if (idx >= (long long)EP * H) return;
    int e = (int)(idx / H), hh = (int)(idx % H);
    int d = (e < EE) ? dst[e] : (e - EE);
    alpha[idx] = alpha[idx] / denom[d * H + hh];
}

// ---------------------------------------------------------------------------
// agg[dst, elem] += w[e, elem>>Cshift] * h[src, elem]
// ---------------------------------------------------------------------------
__global__ void aggregate_kernel(const int* __restrict__ src, const int* __restrict__ dst,
                                 const float* __restrict__ w, const _Float16* __restrict__ h,
                                 float* __restrict__ agg, int Nc, int Cshift, int H) {
    long long total = (long long)EP * Nc;
    long long idx = (long long)blockIdx.x * blockDim.x + threadIdx.x;
    long long stride = (long long)gridDim.x * blockDim.x;
    for (; idx < total; idx += stride) {
        int elem = (int)(idx % Nc);
        long long e = idx / Nc;
        int s, d;
        if (e < EE) { s = src[e]; d = dst[e]; }
        else        { s = d = (int)(e - EE); }
        int hh = elem >> Cshift;
        float val = w[e * H + hh] * (float)h[(size_t)s * Nc + elem];
        atomicAdd(&agg[(size_t)d * Nc + elem], val);
    }
}

// ---------------------------------------------------------------------------
// x = relu(agg + b); keep f32 (for concat) and optional f16 (next layer GEMM)
// ---------------------------------------------------------------------------
__global__ void bias_relu_kernel(const float* __restrict__ agg, const float* __restrict__ b,
                                 float* __restrict__ xf, _Float16* __restrict__ xh, int Nc) {
    long long idx = (long long)blockIdx.x * blockDim.x + threadIdx.x;
    if (idx >= (long long)NN * Nc) return;
    int j = (int)(idx % Nc);
    float v = agg[idx] + b[j];
    v = (v > 0.0f) ? v : 0.0f;
    xf[idx] = v;
    if (xh) xh[idx] = (_Float16)v;
}

// ---------------------------------------------------------------------------
// out[n] = sigmoid(concat(x1,x2,x3)[n] . Wf + bf) — one wave per node
// ---------------------------------------------------------------------------
__global__ __launch_bounds__(256)
void final_dot_kernel(const float* __restrict__ x1, const float* __restrict__ x2,
                      const float* __restrict__ x3, const float* __restrict__ Wf,
                      const float* __restrict__ bf, float* __restrict__ out) {
    int wave = (int)((blockIdx.x * blockDim.x + threadIdx.x) >> 5);
    int lane = threadIdx.x & 31;
    if (wave >= NN) return;
    float s = 0.0f;
    for (int j = lane; j < 256; j += 32) s += x1[(size_t)wave * 256 + j] * Wf[j];
    for (int j = lane; j < 256; j += 32) s += x2[(size_t)wave * 256 + j] * Wf[256 + j];
    for (int j = lane; j < 768; j += 32) s += x3[(size_t)wave * 768 + j] * Wf[512 + j];
#pragma unroll
    for (int off = 16; off; off >>= 1) s += __shfl_xor(s, off, 32);
    if (lane == 0) out[wave] = 1.0f / (1.0f + expf(-(s + bf[0])));
}

// ---------------------------------------------------------------------------
// Host launcher
// ---------------------------------------------------------------------------
extern "C" void kernel_launch(void* const* d_in, const int* in_sizes, int n_in,
                              void* d_out, int out_size, void* d_ws, size_t ws_size,
                              hipStream_t stream) {
    (void)in_sizes; (void)n_in; (void)out_size; (void)ws_size;

    const float* x   = (const float*)d_in[0];
    const int*   ei  = (const int*)d_in[1];
    const float* ea  = (const float*)d_in[2];
    const int* src = ei;
    const int* dst = ei + EE;

    const float* W[3]    = { (const float*)d_in[3],  (const float*)d_in[9],  (const float*)d_in[15] };
    const float* atts[3] = { (const float*)d_in[4],  (const float*)d_in[10], (const float*)d_in[16] };
    const float* attd[3] = { (const float*)d_in[5],  (const float*)d_in[11], (const float*)d_in[17] };
    const float* We[3]   = { (const float*)d_in[6],  (const float*)d_in[12], (const float*)d_in[18] };
    const float* atte[3] = { (const float*)d_in[7],  (const float*)d_in[13], (const float*)d_in[19] };
    const float* bb[3]   = { (const float*)d_in[8],  (const float*)d_in[14], (const float*)d_in[20] };
    const float* Wf = (const float*)d_in[21];
    const float* bf = (const float*)d_in[22];
    float* out = (float*)d_out;

    // ---- workspace carve-up (256B aligned) ----
    char* base = (char*)d_ws;
    size_t off = 0;
    auto take = [&](size_t bytes) -> void* {
        void* p = base + off;
        off += (bytes + 255) & ~(size_t)255;
        return p;
    };
    float*    cnt   = (float*)take((size_t)NN * 4);
    float*    mea   = (float*)take((size_t)NN * EDIM * 4);
    _Float16* xh0   = (_Float16*)take((size_t)NN * 32  * 2);
    _Float16* xh1   = (_Float16*)take((size_t)NN * 256 * 2);
    _Float16* xh2   = (_Float16*)take((size_t)NN * 256 * 2);
    _Float16* hbuf  = (_Float16*)take((size_t)NN * 768 * 2);
    float*    agg   = (float*)take((size_t)NN * 768 * 4);
    float*    x1    = (float*)take((size_t)NN * 256 * 4);
    float*    x2    = (float*)take((size_t)NN * 256 * 4);
    float*    x3    = (float*)take((size_t)NN * 768 * 4);
    float*    alpha = (float*)take((size_t)EP * 12 * 4);
    float*    amax  = (float*)take((size_t)NN * 12 * 4);
    float*    denom = (float*)take((size_t)NN * 12 * 4);
    float*    asrc  = (float*)take((size_t)NN * 12 * 4);
    float*    adst  = (float*)take((size_t)NN * 12 * 4);
    _Float16* wpack = (_Float16*)take((size_t)256 * 768 * 2);
    float*    wea   = (float*)take((size_t)EDIM * 12 * 4);

    auto cdiv = [](long long a, long long b) -> int { return (int)((a + b - 1) / b); };
    auto fill = [&](float* p, float v, long long n) {
        int blocks = cdiv(n, 256); if (blocks > (1 << 20)) blocks = 1 << 20;
        hipLaunchKernelGGL(fill_f32, dim3(blocks), dim3(256), 0, stream, p, v, n);
    };

    // ---- self-loop mean edge attrs ----
    fill(cnt, 0.0f, NN);
    fill(mea, 0.0f, (long long)NN * EDIM);
    hipLaunchKernelGGL(loop_cnt_kernel,   dim3(cdiv(EE, 256)), dim3(256), 0, stream, dst, cnt);
    hipLaunchKernelGGL(loop_easum_kernel, dim3(cdiv((long long)EE * EDIM, 256)), dim3(256), 0, stream, dst, ea, mea);
    hipLaunchKernelGGL(loop_mean_kernel,  dim3(cdiv((long long)NN * EDIM, 256)), dim3(256), 0, stream, cnt, mea);

    // ---- input f16 conversion (K padded 16 -> 32) ----
    hipLaunchKernelGGL(convert_pad_x_kernel, dim3(cdiv((long long)NN * 32, 256)), dim3(256), 0, stream, x, xh0);

    const float neg_inf = -__builtin_huge_valf();

    auto run_layer = [&](const _Float16* xh_in, int Kpad, int Kin, int li,
                         int H, int C, float* xf_out, _Float16* xh_out) {
        const int Nc = H * C;
        const int Cshift = (C == 32) ? 5 : 6;

        // pack weights into WMMA B layout
        long long ptot = (long long)(Kpad >> 5) * (Nc >> 4) * 512;
        hipLaunchKernelGGL(pack_w_kernel, dim3(cdiv(ptot, 256)), dim3(256), 0, stream,
                           W[li], wpack, Kin, Kpad, Nc);

        // h = xh_in @ W  (f16 in, f32 acc, f16 out)
        hipLaunchKernelGGL(gemm_wmma_f16, dim3(NN / 16, Nc / 16), dim3(32), 0, stream,
                           xh_in, wpack, hbuf, Kpad, Nc);

        // attention scalars
        hipLaunchKernelGGL(attn_node_kernel, dim3(cdiv((long long)NN * H, 256)), dim3(256), 0, stream,
                           hbuf, atts[li], attd[li], asrc, adst, H, C);
        hipLaunchKernelGGL(fold_we_kernel, dim3(cdiv(EDIM * H, 256)), dim3(256), 0, stream,
                           We[li], atte[li], wea, H, C);

        fill(amax, neg_inf, (long long)NN * H);
        fill(denom, 0.0f,   (long long)NN * H);
        fill(agg,  0.0f,    (long long)NN * Nc);

        int eblocks = cdiv((long long)EP * H, 256);
        hipLaunchKernelGGL(edge_alpha_kernel, dim3(eblocks), dim3(256), 0, stream,
                           src, dst, ea, mea, wea, asrc, adst, alpha, amax, H);
        hipLaunchKernelGGL(edge_exp_kernel,  dim3(eblocks), dim3(256), 0, stream,
                           dst, amax, alpha, denom, H);
        hipLaunchKernelGGL(edge_norm_kernel, dim3(eblocks), dim3(256), 0, stream,
                           dst, denom, alpha, H);

        long long atot = (long long)EP * Nc;
        int ablocks = cdiv(atot, 256); if (ablocks > (1 << 20)) ablocks = 1 << 20;
        hipLaunchKernelGGL(aggregate_kernel, dim3(ablocks), dim3(256), 0, stream,
                           src, dst, alpha, hbuf, agg, Nc, Cshift, H);

        hipLaunchKernelGGL(bias_relu_kernel, dim3(cdiv((long long)NN * Nc, 256)), dim3(256), 0, stream,
                           agg, bb[li], xf_out, xh_out, Nc);
    };

    run_layer(xh0, 32,  16,  0,  8, 32, x1, xh1);
    run_layer(xh1, 256, 256, 1,  8, 32, x2, xh2);
    run_layer(xh2, 256, 256, 2, 12, 64, x3, nullptr);

    hipLaunchKernelGGL(final_dot_kernel, dim3(cdiv((long long)NN * 32, 256)), dim3(256), 0, stream,
                       x1, x2, x3, Wf, bf, out);
}
